// Model1_81432579932506
// MI455X (gfx1250) — matrix operand order repaired and checked by reading, not compiled, not run on hardware
//
#include <hip/hip_runtime.h>

typedef __attribute__((ext_vector_type(16))) __bf16          v16bf;
typedef __attribute__((ext_vector_type(8)))  float           v8f;
typedef __attribute__((ext_vector_type(8)))  unsigned short  us8;
typedef __attribute__((ext_vector_type(4)))  float           f4;

#define CN    65536      // nodes
#define CE    262144     // edges
#define CB    2048       // graphs / batch
#define CL    30         // protein seq len
#define CFIN  64
#define CH    256
#define CP    768
#define CP1   384
#define CP2   192
#define CKW   10
#define CPAD  8
#define CL1   37
#define CL2   28
#define CFC1  5632       // 256 + 192*28
#define CEPS  1e-5f

// Native f32 -> bf16 conversion (fptrunc, RNE) -> raw bits
__device__ __forceinline__ unsigned short bf16bits(float f) {
  __bf16 h = (__bf16)f;
  return __builtin_bit_cast(unsigned short, h);
}
__device__ __forceinline__ us8 pack8(f4 a, f4 b) {
  union { us8 u; __bf16 h[8]; } r;
  r.h[0] = (__bf16)a[0]; r.h[1] = (__bf16)a[1];
  r.h[2] = (__bf16)a[2]; r.h[3] = (__bf16)a[3];
  r.h[4] = (__bf16)b[0]; r.h[5] = (__bf16)b[1];
  r.h[6] = (__bf16)b[2]; r.h[7] = (__bf16)b[3];
  return r.u;
}

union Frag { us8 h[2]; v16bf b; };

// ---------------------------------------------------------------------------
// LDS-free bf16 WMMA GEMM. Block = 128 threads (4 waves), tile 128M x 64N,
// K step 32. Each wave owns 32 M-rows; 8 accumulators; 8 WMMAs / K-step.
// B operand is pre-packed K-contiguous: B[kt][n][kk] (kt=K/32, kk=K%32) so a
// lane's fragment = two aligned b128 loads. A fragments are loaded straight
// from global (A is K-contiguous in every mode).
//   MODE 0: A f32 [M,K], C f32 [M,N]
//   MODE 1: A f32, store relu(acc + bias[n])
//   MODE 2: A = implicit im2col of protein [B,30,768] (pad 8), row m=(b,l1);
//           store bf16 relu(acc+bias[n]) to out1[m*384+n]
//   MODE 3: A = implicit im2col of out1 bf16 [B,37,384], row m=(b,l2);
//           store relu(acc+bias[n]) to z[b, 256 + n*28 + l2]
// ---------------------------------------------------------------------------
struct GemmArgs {
  const float*          A;
  const unsigned short* B;     // packed bf16 [K/32][N][32]
  float*                C;
  unsigned short*       Cb;
  const float*          bias;
  const float*          prot;  // MODE 2
  const unsigned short* Ab;    // MODE 3
  int M, N, K, lda, ldc;
};

template <int MODE>
__global__ __launch_bounds__(128) void wmma_gemm(GemmArgs g) {
  const int tid  = threadIdx.x;
  const int wv   = tid >> 5;
  const int lane = tid & 31;
  const int hf   = lane >> 4;
  const int l15  = lane & 15;

  const int m0 = blockIdx.y * 128;
  const int n0 = blockIdx.x * 64;
  const int mA = m0 + (wv << 5) + l15;   // lane's first A row; second = mA+16

  int pb0 = 0, pl0 = 0, pb1 = 0, pl1 = 0;
  if constexpr (MODE == 2 || MODE == 3) {
    const int ld = (MODE == 2) ? CL1 : CL2;
    pb0 = mA / ld;        pl0 = mA - pb0 * ld;
    pb1 = (mA + 16) / ld; pl1 = (mA + 16) - pb1 * ld;
  }

  const v8f vz = {0.f, 0.f, 0.f, 0.f, 0.f, 0.f, 0.f, 0.f};
  v8f acc[2][4];
#pragma unroll
  for (int h = 0; h < 2; ++h)
#pragma unroll
    for (int ns = 0; ns < 4; ++ns) acc[h][ns] = vz;

  int tap = 0, cb = 0;                   // conv-mode K decomposition
  const int ksteps = g.K >> 5;
  for (int ks = 0; ks < ksteps; ++ks) {
    const int k0 = ks << 5;

    // ---- A fragments: u[0..7] = K k0+hf*8.., u[8..15] = K k0+16+hf*8.. ----
    Frag a0, a1;
    if constexpr (MODE == 0 || MODE == 1) {
      const float* ap0 = g.A + (long)mA * g.lda + k0 + (hf << 3);
      const float* ap1 = ap0 + (long)16 * g.lda;
      a0.h[0] = pack8(*(const f4*)ap0,        *(const f4*)(ap0 + 4));
      a0.h[1] = pack8(*(const f4*)(ap0 + 16), *(const f4*)(ap0 + 20));
      a1.h[0] = pack8(*(const f4*)ap1,        *(const f4*)(ap1 + 4));
      a1.h[1] = pack8(*(const f4*)(ap1 + 16), *(const f4*)(ap1 + 20));
    } else if constexpr (MODE == 2) {
      const f4 z4 = {0.f, 0.f, 0.f, 0.f};
      const int l0 = pl0 + tap - CPAD;
      const int l1 = pl1 + tap - CPAD;
      f4 t0 = z4, t1 = z4, t2 = z4, t3 = z4;
      if (l0 >= 0 && l0 < CL) {
        const float* p = g.prot + ((long)pb0 * CL + l0) * CP + cb + (hf << 3);
        t0 = *(const f4*)p;        t1 = *(const f4*)(p + 4);
        t2 = *(const f4*)(p + 16); t3 = *(const f4*)(p + 20);
      }
      a0.h[0] = pack8(t0, t1); a0.h[1] = pack8(t2, t3);
      t0 = z4; t1 = z4; t2 = z4; t3 = z4;
      if (l1 >= 0 && l1 < CL) {
        const float* p = g.prot + ((long)pb1 * CL + l1) * CP + cb + (hf << 3);
        t0 = *(const f4*)p;        t1 = *(const f4*)(p + 4);
        t2 = *(const f4*)(p + 16); t3 = *(const f4*)(p + 20);
      }
      a1.h[0] = pack8(t0, t1); a1.h[1] = pack8(t2, t3);
    } else {  // MODE 3: bf16 operand, direct b128 loads
      const unsigned short* p0 =
          g.Ab + ((long)pb0 * CL1 + pl0 + tap) * CP1 + cb + (hf << 3);
      const unsigned short* p1 =
          g.Ab + ((long)pb1 * CL1 + pl1 + tap) * CP1 + cb + (hf << 3);
      a0.h[0] = *(const us8*)p0; a0.h[1] = *(const us8*)(p0 + 16);
      a1.h[0] = *(const us8*)p1; a1.h[1] = *(const us8*)(p1 + 16);
    }

    // ---- B fragments (K-contiguous packed) + 8 WMMAs ----
    const unsigned short* bbase =
        g.B + (((long)ks * g.N + n0 + l15) << 5) + (hf << 3);
#pragma unroll
    for (int ns = 0; ns < 4; ++ns) {
      Frag bf;
      const unsigned short* bp = bbase + ((long)ns << 9);  // +16 cols * 32
      bf.h[0] = *(const us8*)bp;
      bf.h[1] = *(const us8*)(bp + 16);
      acc[0][ns] = __builtin_amdgcn_wmma_f32_16x16x32_bf16(
          false, a0.b, false, bf.b, (short)0, acc[0][ns], false, false);
      acc[1][ns] = __builtin_amdgcn_wmma_f32_16x16x32_bf16(
          false, a1.b, false, bf.b, (short)0, acc[1][ns], false, false);
    }

    if constexpr (MODE == 2) { cb += 32; if (cb == CP)  { cb = 0; ++tap; } }
    if constexpr (MODE == 3) { cb += 32; if (cb == CP1) { cb = 0; ++tap; } }
  }

  // ---- store (C/D: lane<16 -> M=v, lane>=16 -> M=v+8; N = lane&15) ----
#pragma unroll
  for (int h = 0; h < 2; ++h) {
#pragma unroll
    for (int ns = 0; ns < 4; ++ns) {
      const int n = n0 + (ns << 4) + l15;
#pragma unroll
      for (int v = 0; v < 8; ++v) {
        const int m = m0 + (wv << 5) + (h << 4) + (hf << 3) + v;
        float r = acc[h][ns][v];
        if constexpr (MODE == 0) {
          g.C[(long)m * g.ldc + n] = r;
        } else if constexpr (MODE == 1) {
          r = fmaxf(r + g.bias[n], 0.f);
          g.C[(long)m * g.ldc + n] = r;
        } else if constexpr (MODE == 2) {
          r = fmaxf(r + g.bias[n], 0.f);
          g.Cb[(long)m * CP1 + n] = bf16bits(r);
        } else {
          r = fmaxf(r + g.bias[n], 0.f);
          int b = m / CL2, l = m - b * CL2;
          g.C[(long)b * CFC1 + CH + n * CL2 + l] = r;
        }
      }
    }
  }
}

// ---------------------------------------------------------------------------
// Weight packing into K-contiguous WMMA layout: out[k/32][n][k%32]
// ---------------------------------------------------------------------------
__global__ void k_pack_kc(const float* w, unsigned short* out, int K, int N) {
  int i = blockIdx.x * blockDim.x + threadIdx.x;
  if (i >= K * N) return;
  int k = i / N, n = i - k * N;
  out[(((long)(k >> 5) * N + n) << 5) + (k & 31)] = bf16bits(w[i]);
}
// conv weight w[o][c][t] f32; GEMM K index = t*C + c, col = o
__global__ void k_pack_conv_kc(const float* w, unsigned short* out, int O, int C,
                               int T) {
  int i = blockIdx.x * blockDim.x + threadIdx.x;
  if (i >= O * C * T) return;
  int o = i / (C * T);
  int rem = i - o * (C * T);
  int c = rem / T;
  int t = rem - c * T;
  int k = t * C + c;
  out[(((long)(k >> 5) * O + o) << 5) + (k & 31)] = bf16bits(w[i]);
}

// ---------------------------------------------------------------------------
// Support kernels
// ---------------------------------------------------------------------------
__global__ void k_zero(float* p, int n) {
  int i = blockIdx.x * blockDim.x + threadIdx.x;
  if (i < n) p[i] = 0.f;
}
__global__ void k_fill1(float* p, int n) {
  int i = blockIdx.x * blockDim.x + threadIdx.x;
  if (i < n) p[i] = 1.f;
}
__global__ void k_deg_count(const int* ei, float* deg, int e) {
  int i = blockIdx.x * blockDim.x + threadIdx.x;
  if (i < e) atomicAdd(&deg[ei[CE + i]], 1.f);
}
__global__ void k_rsqrt(float* p, int n) {
  int i = blockIdx.x * blockDim.x + threadIdx.x;
  if (i < n) p[i] = rsqrtf(p[i]);
}
__global__ void k_scatter_self(const float* xw, const float* dis, float* out) {
  int i = blockIdx.x * blockDim.x + threadIdx.x;  // N*H threads
  int row = i >> 8;
  float s = dis[row];
  out[i] = xw[i] * s * s;
}
__global__ void k_scatter_edges(const int* ei, const float* xw, const float* dis,
                                float* out) {
  int t = blockIdx.x * blockDim.x + threadIdx.x;  // E*64 threads, 4 feats each
  int e = t >> 6;
  int f = (t & 63) << 2;
  int src = ei[e];
  int dst = ei[CE + e];
  float nm = dis[src] * dis[dst];
  const float* xr = xw + (long)src * CH + f;
  float* orow = out + (long)dst * CH + f;
  atomicAdd(orow + 0, xr[0] * nm);
  atomicAdd(orow + 1, xr[1] * nm);
  atomicAdd(orow + 2, xr[2] * nm);
  atomicAdd(orow + 3, xr[3] * nm);
}
__global__ void k_bn_reduce(const float* h, float* sums, int rowsPerBlock) {
  int f = threadIdx.x;  // 256 threads = one per feature, coalesced row reads
  long base = (long)blockIdx.x * rowsPerBlock;
  float s = 0.f, s2 = 0.f;
  for (int r = 0; r < rowsPerBlock; ++r) {
    float v = h[(base + r) * CH + f];
    s += v;
    s2 += v * v;
  }
  atomicAdd(&sums[f], s);
  atomicAdd(&sums[CH + f], s2);
}
__global__ void k_bn_apply(float* h, const float* sums, const float* g,
                           const float* b) {
  int i = blockIdx.x * blockDim.x + threadIdx.x;  // N*H
  int f = i & (CH - 1);
  float m = sums[f] * (1.f / CN);
  float v = sums[CH + f] * (1.f / CN) - m * m;
  float y = (h[i] - m) * rsqrtf(v + CEPS) * g[f] + b[f];
  h[i] = fmaxf(y, 0.f);
}
__global__ void k_pool_accum(const float* h, const int* batch, float* z,
                             float* cnt) {
  int i = blockIdx.x * blockDim.x + threadIdx.x;  // N*H
  int row = i >> 8;
  int f = i & (CH - 1);
  int b = batch[row];
  atomicAdd(&z[(long)b * CFC1 + f], h[i]);
  if (f == 0) atomicAdd(&cnt[b], 1.f);
}
__global__ void k_pool_div(float* z, const float* cnt) {
  int i = blockIdx.x * blockDim.x + threadIdx.x;  // B*H
  int b = i >> 8;
  int f = i & (CH - 1);
  z[(long)b * CFC1 + f] /= fmaxf(cnt[b], 1.f);
}
__global__ void k_out_head(const float* a, const float* w, const float* bia,
                           float* out) {
  int i = blockIdx.x * blockDim.x + threadIdx.x;
  if (i >= CB) return;
  float acc = bia[0];
  const float* r = a + i * 64;
#pragma unroll
  for (int j = 0; j < 64; ++j) acc += r[j] * w[j];
  out[i] = 1.f / (1.f + expf(-acc));
}

// ---------------------------------------------------------------------------
extern "C" void kernel_launch(void* const* d_in, const int* in_sizes, int n_in,
                              void* d_out, int out_size, void* d_ws, size_t ws_size,
                              hipStream_t stream) {
  (void)in_sizes; (void)n_in; (void)out_size; (void)ws_size;
  const float* x       = (const float*)d_in[0];
  const int*   ei      = (const int*)d_in[1];
  const int*   batch   = (const int*)d_in[2];
  const float* protein = (const float*)d_in[3];
  const float* gcn1_w  = (const float*)d_in[4];
  // gcn biases (d_in[5], d_in[7]) cancel through batchnorm mean subtraction
  const float* gcn2_w  = (const float*)d_in[6];
  const float* bn1_g   = (const float*)d_in[8];
  const float* bn1_b   = (const float*)d_in[9];
  const float* bn2_g   = (const float*)d_in[10];
  const float* bn2_b   = (const float*)d_in[11];
  const float* pc1_w   = (const float*)d_in[12];
  const float* pc1_b   = (const float*)d_in[13];
  const float* pc2_w   = (const float*)d_in[14];
  const float* pc2_b   = (const float*)d_in[15];
  const float* fc1_w   = (const float*)d_in[16];
  const float* fc1_b   = (const float*)d_in[17];
  const float* fc2_w   = (const float*)d_in[18];
  const float* fc2_b   = (const float*)d_in[19];
  const float* fc3_w   = (const float*)d_in[20];
  const float* fc3_b   = (const float*)d_in[21];
  const float* out_w   = (const float*)d_in[22];
  const float* out_b   = (const float*)d_in[23];
  float* out = (float*)d_out;

  char* ws = (char*)d_ws;
  size_t off = 0;
  auto alloc = [&](size_t bytes) -> char* {
    char* p = ws + off;
    off += (bytes + 255) & ~(size_t)255;
    return p;
  };
  float* deg           = (float*)alloc((size_t)CN * 4);
  float* bufA          = (float*)alloc((size_t)CN * CH * 4);
  float* bufB          = (float*)alloc((size_t)CN * CH * 4);
  float* sums          = (float*)alloc(2 * CH * 4);
  float* cnt           = (float*)alloc((size_t)CB * 4);
  unsigned short* out1 = (unsigned short*)alloc((size_t)CB * CL1 * CP1 * 2);
  float* z             = (float*)alloc((size_t)CB * CFC1 * 4);
  float* a1            = (float*)alloc((size_t)CB * 256 * 4);
  float* a2            = (float*)alloc((size_t)CB * 128 * 4);
  float* a3            = (float*)alloc((size_t)CB * 64 * 4);
  unsigned short* w1p  = (unsigned short*)alloc((size_t)CFIN * CH * 2);
  unsigned short* w2p  = (unsigned short*)alloc((size_t)CH * CH * 2);
  unsigned short* pc1p = (unsigned short*)alloc((size_t)CP * CKW * CP1 * 2);
  unsigned short* pc2p = (unsigned short*)alloc((size_t)CP1 * CKW * CP2 * 2);
  unsigned short* fc1p = (unsigned short*)alloc((size_t)CFC1 * 256 * 2);
  unsigned short* fc2p = (unsigned short*)alloc((size_t)256 * 128 * 2);
  unsigned short* fc3p = (unsigned short*)alloc((size_t)128 * 64 * 2);

  auto grid1 = [](int n) { return dim3((unsigned)((n + 255) / 256)); };

  // ---- pack weights to bf16 K-contiguous layout ----
  k_pack_kc<<<grid1(CFIN * CH), 256, 0, stream>>>(gcn1_w, w1p, CFIN, CH);
  k_pack_kc<<<grid1(CH * CH), 256, 0, stream>>>(gcn2_w, w2p, CH, CH);
  k_pack_kc<<<grid1(CFC1 * 256), 256, 0, stream>>>(fc1_w, fc1p, CFC1, 256);
  k_pack_kc<<<grid1(256 * 128), 256, 0, stream>>>(fc2_w, fc2p, 256, 128);
  k_pack_kc<<<grid1(128 * 64), 256, 0, stream>>>(fc3_w, fc3p, 128, 64);
  k_pack_conv_kc<<<grid1(CP1 * CP * CKW), 256, 0, stream>>>(pc1_w, pc1p, CP1, CP, CKW);
  k_pack_conv_kc<<<grid1(CP2 * CP1 * CKW), 256, 0, stream>>>(pc2_w, pc2p, CP2, CP1, CKW);

  // ---- symmetric GCN normalization: dis = rsqrt(1 + in-degree) ----
  k_fill1<<<grid1(CN), 256, 0, stream>>>(deg, CN);
  k_deg_count<<<grid1(CE), 256, 0, stream>>>(ei, deg, CE);
  k_rsqrt<<<grid1(CN), 256, 0, stream>>>(deg, CN);

  // ---- GCN layer 1: (x @ w1) -> scatter -> BN+ReLU ----
  GemmArgs g1{};
  g1.A = x; g1.B = w1p; g1.C = bufA;
  g1.M = CN; g1.N = CH; g1.K = CFIN; g1.lda = CFIN; g1.ldc = CH;
  wmma_gemm<0><<<dim3(CH / 64, CN / 128), 128, 0, stream>>>(g1);
  k_scatter_self<<<grid1(CN * CH), 256, 0, stream>>>(bufA, deg, bufB);
  k_scatter_edges<<<grid1(CE * 64), 256, 0, stream>>>(ei, bufA, deg, bufB);
  k_zero<<<grid1(2 * CH), 256, 0, stream>>>(sums, 2 * CH);
  k_bn_reduce<<<dim3(CN / 64), 256, 0, stream>>>(bufB, sums, 64);
  k_bn_apply<<<grid1(CN * CH), 256, 0, stream>>>(bufB, sums, bn1_g, bn1_b);

  // ---- GCN layer 2 ----
  GemmArgs g2{};
  g2.A = bufB; g2.B = w2p; g2.C = bufA;
  g2.M = CN; g2.N = CH; g2.K = CH; g2.lda = CH; g2.ldc = CH;
  wmma_gemm<0><<<dim3(CH / 64, CN / 128), 128, 0, stream>>>(g2);
  k_scatter_self<<<grid1(CN * CH), 256, 0, stream>>>(bufA, deg, bufB);
  k_scatter_edges<<<grid1(CE * 64), 256, 0, stream>>>(ei, bufA, deg, bufB);
  k_zero<<<grid1(2 * CH), 256, 0, stream>>>(sums, 2 * CH);
  k_bn_reduce<<<dim3(CN / 64), 256, 0, stream>>>(bufB, sums, 64);
  k_bn_apply<<<grid1(CN * CH), 256, 0, stream>>>(bufB, sums, bn2_g, bn2_b);

  // ---- global mean pool into z[:, 0:256] ----
  k_zero<<<grid1(CB * CFC1), 256, 0, stream>>>(z, CB * CFC1);
  k_zero<<<grid1(CB), 256, 0, stream>>>(cnt, CB);
  k_pool_accum<<<grid1(CN * CH), 256, 0, stream>>>(bufB, batch, z, cnt);
  k_pool_div<<<grid1(CB * CH), 256, 0, stream>>>(z, cnt);

  // ---- protein conv1 (implicit im2col WMMA GEMM) -> out1 bf16 ----
  GemmArgs c1{};
  c1.B = pc1p; c1.Cb = out1; c1.bias = pc1_b; c1.prot = protein;
  c1.M = CB * CL1; c1.N = CP1; c1.K = CP * CKW;
  wmma_gemm<2><<<dim3(CP1 / 64, (CB * CL1) / 128), 128, 0, stream>>>(c1);

  // ---- protein conv2 -> z[:, 256:] (channel-major flatten) ----
  GemmArgs c2{};
  c2.B = pc2p; c2.C = z; c2.bias = pc2_b; c2.Ab = out1;
  c2.M = CB * CL2; c2.N = CP2; c2.K = CP1 * CKW;
  wmma_gemm<3><<<dim3(CP2 / 64, (CB * CL2) / 128), 128, 0, stream>>>(c2);

  // ---- FC stack ----
  GemmArgs f1{};
  f1.A = z; f1.B = fc1p; f1.C = a1; f1.bias = fc1_b;
  f1.M = CB; f1.N = 256; f1.K = CFC1; f1.lda = CFC1; f1.ldc = 256;
  wmma_gemm<1><<<dim3(256 / 64, CB / 128), 128, 0, stream>>>(f1);
  GemmArgs f2{};
  f2.A = a1; f2.B = fc2p; f2.C = a2; f2.bias = fc2_b;
  f2.M = CB; f2.N = 128; f2.K = 256; f2.lda = 256; f2.ldc = 128;
  wmma_gemm<1><<<dim3(128 / 64, CB / 128), 128, 0, stream>>>(f2);
  GemmArgs f3{};
  f3.A = a2; f3.B = fc3p; f3.C = a3; f3.bias = fc3_b;
  f3.M = CB; f3.N = 64; f3.K = 128; f3.lda = 128; f3.ldc = 64;
  wmma_gemm<1><<<dim3(64 / 64, CB / 128), 128, 0, stream>>>(f3);

  k_out_head<<<grid1(CB), 256, 0, stream>>>(a3, out_w, out_b, out);
}
